// MultiHeadAttentionBlock_26018911879838
// MI455X (gfx1250) — matrix operand order repaired and checked by reading, not compiled, not run on hardware
//
#include <hip/hip_runtime.h>

#define S_LEN 2048
#define D_DIM 1024
#define H_NUM 16
#define DK_DIM 64
#define B_NUM 2

typedef __attribute__((ext_vector_type(16))) __bf16        v16bf;
typedef __attribute__((ext_vector_type(8)))  float         v8f;
typedef __attribute__((ext_vector_type(4)))  unsigned int  u32x4;
typedef __attribute__((ext_vector_type(2)))  unsigned int  u32x2;
typedef __attribute__((ext_vector_type(4)))  float         f32x4;
typedef __attribute__((__vector_size__(16))) int           i32x4v;   // builtin's pointee type
typedef unsigned short u16;

union ABu {
    v16bf v;
    u32x4 u[2];
    u16   h[16];
};

// fp32 -> bf16 via native hardware conversion (v_cvt_pk_bf16_f32 class).
__device__ __forceinline__ u16 f2bf(float f) {
    union { __bf16 b; u16 s; } u;
    u.b = (__bf16)f;
    return u.s;
}

__device__ __forceinline__ v8f vzero() {
    v8f z = {0.f, 0.f, 0.f, 0.f, 0.f, 0.f, 0.f, 0.f};
    return z;
}

__device__ __forceinline__ v8f wmma_bf16(v16bf a, v16bf b, v8f c) {
    // D = A(16x32 bf16) x B(32x16 bf16) + C(16x16 f32)
    return __builtin_amdgcn_wmma_f32_16x16x32_bf16(false, a, false, b,
                                                   (short)0, c, false, false);
}

// Async global->LDS 16-byte copy (per-lane), tracked by ASYNCcnt.
__device__ __forceinline__ void async_b128(const void* g, void* l) {
#if __has_builtin(__builtin_amdgcn_global_load_async_to_lds_b128)
    __builtin_amdgcn_global_load_async_to_lds_b128(
        (__attribute__((address_space(1))) i32x4v*)(uintptr_t)g,
        (__attribute__((address_space(3))) i32x4v*)(unsigned)(uintptr_t)l,
        0, 0);
#else
    unsigned lo = (unsigned)(uintptr_t)l;
    asm volatile("global_load_async_to_lds_b128 %0, %1, off"
                 :: "v"(lo), "v"(g) : "memory");
#endif
}

__device__ __forceinline__ void wait_async0() {
    asm volatile("s_wait_asynccnt 0x0" ::: "memory");
}

// ---------------------------------------------------------------------------
// Kernel 1: Y = X @ Wq.T for X in {q,k,v} (blockIdx.z = mode).
// Block: 4 waves = 64 rows x 64 cols. W tile (64x32) staged once per block in
// LDS as bf16 (converted once, shared by all 4 waves), double buffered.
// mode 0 (Q): store * 0.125 into [BH][S][DK]
// mode 1 (K): store into [BH][S][DK]
// mode 2 (V): store transposed into [BH][DK][S]
// ---------------------------------------------------------------------------
__global__ __launch_bounds__(128)
void proj_qkv_wmma(const float* __restrict__ q, const float* __restrict__ k,
                   const float* __restrict__ v, const float* __restrict__ W,
                   u16* __restrict__ ws) {
    __shared__ __align__(16) u16 wsh[2][64 * 32];

    const int tid  = threadIdx.x;
    const int lane = tid & 31;
    const int wv   = tid >> 5;
    const int half = lane >> 4;
    const int ln   = lane & 15;
    const int mode = blockIdx.z;

    const float* X = (mode == 0) ? q : (mode == 1) ? k : v;
    u16* Y = ws + (size_t)mode * (size_t)(B_NUM * S_LEN * D_DIM);
    const float scale = (mode == 0) ? 0.125f : 1.0f;   // 1/sqrt(DK) folded into Q

    const int m0 = blockIdx.x * 64 + wv * 16;
    const int n0 = blockIdx.y * 64;

    // Cooperative stage of W[n0..n0+63][k0..k0+31] -> bf16 LDS tile [64][32].
    auto stageW = [&](int buf, int k0) {
#pragma unroll
        for (int i = 0; i < 4; ++i) {
            int c   = tid + i * 128;        // 0..511
            int row = c >> 3, o = c & 7;    // row 0..63, o 0..7 (4 floats each)
            f32x4 g = *(const f32x4*)(W + (size_t)(n0 + row) * D_DIM + k0 + o * 4);
            unsigned p0 = (unsigned)f2bf(g[0]) | ((unsigned)f2bf(g[1]) << 16);
            unsigned p1 = (unsigned)f2bf(g[2]) | ((unsigned)f2bf(g[3]) << 16);
            u32x2 pk = {p0, p1};
            *(u32x2*)&wsh[buf][row * 32 + o * 4] = pk;
        }
    };

    v8f acc[4];
    acc[0] = vzero(); acc[1] = vzero(); acc[2] = vzero(); acc[3] = vzero();

    const float* arow = X + (size_t)(m0 + ln) * D_DIM;

    stageW(0, 0);
    __syncthreads();
    int buf = 0;
    for (int k0 = 0; k0 < D_DIM; k0 += 32) {
        if (k0 + 32 < D_DIM) stageW(buf ^ 1, k0 + 32);

        // A operand: lane(half, m=ln); vec[0..7]=X[m][k0+8h..], vec[8..15]=X[m][k0+16+8h..]
        ABu a;
        {
            f32x4 f0 = *(const f32x4*)(arow + k0 + 8 * half);
            f32x4 f1 = *(const f32x4*)(arow + k0 + 8 * half + 4);
            f32x4 f2 = *(const f32x4*)(arow + k0 + 16 + 8 * half);
            f32x4 f3 = *(const f32x4*)(arow + k0 + 16 + 8 * half + 4);
#pragma unroll
            for (int j = 0; j < 4; ++j) {
                a.h[j]      = f2bf(f0[j]);
                a.h[4 + j]  = f2bf(f1[j]);
                a.h[8 + j]  = f2bf(f2[j]);
                a.h[12 + j] = f2bf(f3[j]);
            }
        }
#pragma unroll
        for (int nt = 0; nt < 4; ++nt) {
            // B operand from LDS: B[kk][n] = W[n0+n][k0+kk], kk = 8*half + j (j<8), 16 + 8*half + (j-8) (j>=8)
            ABu bm;
            const u16* br = &wsh[buf][(nt * 16 + ln) * 32 + 8 * half];
            bm.u[0] = ((const u32x4*)br)[0];
            bm.u[1] = ((const u32x4*)br)[2];
            acc[nt] = wmma_bf16(a.v, bm.v, acc[nt]);
        }
        __syncthreads();
        buf ^= 1;
    }

    // Store: C layout = lane(half, n=ln), VGPR r -> row r + 8*half
#pragma unroll
    for (int nt = 0; nt < 4; ++nt) {
#pragma unroll
        for (int r = 0; r < 8; ++r) {
            int row = m0 + r + 8 * half;
            int b   = row >> 11;              // / S_LEN
            int s   = row & (S_LEN - 1);
            int col = n0 + nt * 16 + ln;
            int h   = col >> 6;               // / DK_DIM
            int dk  = col & (DK_DIM - 1);
            size_t idx;
            if (mode == 2)
                idx = ((size_t)(b * H_NUM + h) * DK_DIM + dk) * S_LEN + s;   // V transposed
            else
                idx = ((size_t)(b * H_NUM + h) * S_LEN + s) * DK_DIM + dk;
            Y[idx] = f2bf(acc[nt][r] * scale);
        }
    }
}

// ---------------------------------------------------------------------------
// Kernel 2: flash attention. Block = 4 waves sharing one (b,h) and one KV
// stream; each wave owns a 16-query tile. K/V blocks (32 keys) are staged in
// LDS via async global->LDS copies (ASYNCcnt), double buffered.
// ---------------------------------------------------------------------------
__global__ __launch_bounds__(128)
void flash_attn_wmma(const u16* __restrict__ Qb,
                     const u16* __restrict__ Kb,
                     const u16* __restrict__ Vt,
                     const int* __restrict__ mask,
                     u16* __restrict__ AO) {
    __shared__ __align__(16) u16 ksh[2][32 * 64];   // [key][dk]
    __shared__ __align__(16) u16 vsh[2][64 * 32];   // [dk][key]
    __shared__ __align__(16) u16 psh[4][16 * 32];   // per-wave P tile

    const int tid  = threadIdx.x;
    const int lane = tid & 31;
    const int wv   = tid >> 5;
    const int half = lane >> 4;
    const int ln   = lane & 15;

    const int bh = blockIdx.x;            // 0..31
    const int b  = bh >> 4;               // / H_NUM
    const int h  = bh & (H_NUM - 1);
    const int q0 = (blockIdx.y * 4 + wv) * 16;

    const u16* Qp = Qb + (size_t)bh * S_LEN * DK_DIM;
    const u16* Kp = Kb + (size_t)bh * S_LEN * DK_DIM;
    const u16* Vp = Vt + (size_t)bh * DK_DIM * S_LEN;
    u16* pt = psh[wv];

    // Cooperative async stage: K block = contiguous 32x64 bf16 (4KB);
    // V block = 64 rows of 32 contiguous bf16 from the transposed layout.
    auto stageKV = [&](int buf, int kv0) {
        const u16* kg = Kp + (size_t)kv0 * DK_DIM;
#pragma unroll
        for (int i = 0; i < 2; ++i) {
            int c = tid + i * 128;                   // 0..255 chunks of 8 bf16
            async_b128(kg + c * 8, &ksh[buf][c * 8]);
        }
#pragma unroll
        for (int i = 0; i < 2; ++i) {
            int c = tid + i * 128;
            int r = c >> 2, o = c & 3;               // dk row 0..63, 4 chunks/row
            async_b128(Vp + (size_t)r * S_LEN + kv0 + o * 8,
                       &vsh[buf][r * 32 + o * 8]);
        }
    };

    // Q A-operand, both 32-wide K chunks of DK=64 (1/sqrt(DK) already folded).
    ABu aq[2];
#pragma unroll
    for (int kc = 0; kc < 2; ++kc) {
        const u16* qr = Qp + (size_t)(q0 + ln) * DK_DIM + kc * 32;
        aq[kc].u[0] = *(const u32x4*)(qr + 8 * half);
        aq[kc].u[1] = *(const u32x4*)(qr + 16 + 8 * half);
    }

    v8f o[4];
    o[0] = vzero(); o[1] = vzero(); o[2] = vzero(); o[3] = vzero();
    float mrun[8], lrun[8];
#pragma unroll
    for (int r = 0; r < 8; ++r) { mrun[r] = -3.0e38f; lrun[r] = 0.f; }

    stageKV(0, 0);
    wait_async0();
    __syncthreads();

    int buf = 0;
    for (int kv0 = 0; kv0 < S_LEN; kv0 += 32) {
        if (kv0 + 32 < S_LEN) stageKV(buf ^ 1, kv0 + 32);   // prefetch next block

        // ---- S = Q K^T from LDS K tile ----
        v8f sc[2];
#pragma unroll
        for (int nt = 0; nt < 2; ++nt) {
            v8f c = vzero();
#pragma unroll
            for (int kc = 0; kc < 2; ++kc) {
                ABu bm;   // B[kk][n] = K[kv0+16nt+n][kc*32+kk]
                const u16* kr = &ksh[buf][(nt * 16 + ln) * 64 + kc * 32 + 8 * half];
                bm.u[0] = ((const u32x4*)kr)[0];
                bm.u[1] = ((const u32x4*)kr)[2];
                c = wmma_bf16(aq[kc].v, bm.v, c);
            }
            sc[nt] = c;
        }

        // ---- mask: where(mask==0, 1e-9, s) ----
#pragma unroll
        for (int nt = 0; nt < 2; ++nt) {
#pragma unroll
            for (int r = 0; r < 8; ++r) {
                int row = q0 + r + 8 * half;
                int col = kv0 + nt * 16 + ln;
                int mv = mask[(size_t)b * S_LEN * S_LEN + (size_t)row * S_LEN + col];
                sc[nt][r] = (mv == 0) ? 1e-9f : sc[nt][r];
            }
        }

        // ---- online softmax (rows live across a 16-lane half) ----
        float mloc[8];
#pragma unroll
        for (int r = 0; r < 8; ++r) mloc[r] = fmaxf(sc[0][r], sc[1][r]);
        for (int off = 1; off < 16; off <<= 1)
#pragma unroll
            for (int r = 0; r < 8; ++r)
                mloc[r] = fmaxf(mloc[r], __shfl_xor(mloc[r], off, 32));

        float alpha[8];
#pragma unroll
        for (int r = 0; r < 8; ++r) {
            float mn = fmaxf(mrun[r], mloc[r]);
            alpha[r] = __expf(mrun[r] - mn);
            mrun[r]  = mn;
        }
#pragma unroll
        for (int nt = 0; nt < 2; ++nt)
#pragma unroll
            for (int r = 0; r < 8; ++r)
                sc[nt][r] = __expf(sc[nt][r] - mrun[r]);

        float rs[8];
#pragma unroll
        for (int r = 0; r < 8; ++r) rs[r] = sc[0][r] + sc[1][r];
        for (int off = 1; off < 16; off <<= 1)
#pragma unroll
            for (int r = 0; r < 8; ++r) rs[r] += __shfl_xor(rs[r], off, 32);
#pragma unroll
        for (int r = 0; r < 8; ++r) lrun[r] = lrun[r] * alpha[r] + rs[r];

#pragma unroll
        for (int nt = 0; nt < 4; ++nt)
#pragma unroll
            for (int r = 0; r < 8; ++r) o[nt][r] *= alpha[r];

        // ---- P: C layout -> LDS (row-major 16x32 bf16) -> A layout ----
#pragma unroll
        for (int nt = 0; nt < 2; ++nt)
#pragma unroll
            for (int r = 0; r < 8; ++r)
                pt[(r + 8 * half) * 32 + nt * 16 + ln] = f2bf(sc[nt][r]);
        asm volatile("s_wait_dscnt 0x0" ::: "memory");

        ABu ap;
        ap.u[0] = *(const u32x4*)(pt + ln * 32 + 8 * half);
        ap.u[1] = *(const u32x4*)(pt + ln * 32 + 16 + 8 * half);

        // ---- O += P V from LDS V tile: B[kk][n] = Vt[n0+n][kv0+kk] ----
#pragma unroll
        for (int nt = 0; nt < 4; ++nt) {
            ABu bv;
            const u16* vr = &vsh[buf][(nt * 16 + ln) * 32 + 8 * half];
            bv.u[0] = ((const u32x4*)vr)[0];
            bv.u[1] = ((const u32x4*)vr)[2];
            o[nt] = wmma_bf16(ap.v, bv.v, o[nt]);
        }

        wait_async0();      // prefetched block resident
        __syncthreads();    // all waves done with old buffer
        buf ^= 1;
    }

    // ---- epilogue: O /= l, write head-concat layout [B*S][D] ----
    float inv[8];
#pragma unroll
    for (int r = 0; r < 8; ++r) inv[r] = 1.0f / lrun[r];
#pragma unroll
    for (int nt = 0; nt < 4; ++nt) {
#pragma unroll
        for (int r = 0; r < 8; ++r) {
            int row = q0 + r + 8 * half;
            int col = h * DK_DIM + nt * 16 + ln;
            AO[((size_t)b * S_LEN + row) * D_DIM + col] = f2bf(o[nt][r] * inv[r]);
        }
    }
}

// ---------------------------------------------------------------------------
// Kernel 3: out = AO(bf16) @ Wo.T, fp32 result. Same LDS weight staging.
// ---------------------------------------------------------------------------
__global__ __launch_bounds__(128)
void out_proj_wmma(const u16* __restrict__ A,
                   const float* __restrict__ Wo,
                   float* __restrict__ out) {
    __shared__ __align__(16) u16 wsh[2][64 * 32];

    const int tid  = threadIdx.x;
    const int lane = tid & 31;
    const int wv   = tid >> 5;
    const int half = lane >> 4;
    const int ln   = lane & 15;

    const int m0 = blockIdx.x * 64 + wv * 16;
    const int n0 = blockIdx.y * 64;

    auto stageW = [&](int buf, int k0) {
#pragma unroll
        for (int i = 0; i < 4; ++i) {
            int c   = tid + i * 128;
            int row = c >> 3, o = c & 7;
            f32x4 g = *(const f32x4*)(Wo + (size_t)(n0 + row) * D_DIM + k0 + o * 4);
            unsigned p0 = (unsigned)f2bf(g[0]) | ((unsigned)f2bf(g[1]) << 16);
            unsigned p1 = (unsigned)f2bf(g[2]) | ((unsigned)f2bf(g[3]) << 16);
            u32x2 pk = {p0, p1};
            *(u32x2*)&wsh[buf][row * 32 + o * 4] = pk;
        }
    };

    v8f acc[4];
    acc[0] = vzero(); acc[1] = vzero(); acc[2] = vzero(); acc[3] = vzero();

    const u16* arow = A + (size_t)(m0 + ln) * D_DIM;

    stageW(0, 0);
    __syncthreads();
    int buf = 0;
    for (int k0 = 0; k0 < D_DIM; k0 += 32) {
        if (k0 + 32 < D_DIM) stageW(buf ^ 1, k0 + 32);

        ABu a;
        a.u[0] = *(const u32x4*)(arow + k0 + 8 * half);
        a.u[1] = *(const u32x4*)(arow + k0 + 16 + 8 * half);
#pragma unroll
        for (int nt = 0; nt < 4; ++nt) {
            ABu bm;
            const u16* br = &wsh[buf][(nt * 16 + ln) * 32 + 8 * half];
            bm.u[0] = ((const u32x4*)br)[0];
            bm.u[1] = ((const u32x4*)br)[2];
            acc[nt] = wmma_bf16(a.v, bm.v, acc[nt]);
        }
        __syncthreads();
        buf ^= 1;
    }

#pragma unroll
    for (int nt = 0; nt < 4; ++nt) {
#pragma unroll
        for (int r = 0; r < 8; ++r) {
            int row = m0 + r + 8 * half;
            int col = n0 + nt * 16 + ln;
            out[(size_t)row * D_DIM + col] = acc[nt][r];
        }
    }
}

// ---------------------------------------------------------------------------
extern "C" void kernel_launch(void* const* d_in, const int* in_sizes, int n_in,
                              void* d_out, int out_size, void* d_ws, size_t ws_size,
                              hipStream_t stream) {
    (void)in_sizes; (void)n_in; (void)out_size; (void)ws_size;

    // setup_inputs order: x, q, k, v, mask, Wq, Wo  (x unused in forward)
    const float* q  = (const float*)d_in[1];
    const float* k  = (const float*)d_in[2];
    const float* v  = (const float*)d_in[3];
    const int*  msk = (const int*)d_in[4];
    const float* Wq = (const float*)d_in[5];
    const float* Wo = (const float*)d_in[6];
    float* out = (float*)d_out;

    u16* ws = (u16*)d_ws;
    const size_t NB = (size_t)B_NUM * S_LEN * D_DIM;   // 4,194,304 elements
    u16* Qb = ws;             // [BH][S][DK] bf16 (pre-scaled by 0.125)
    u16* Kb = ws + NB;        // [BH][S][DK] bf16
    u16* Vt = ws + 2 * NB;    // [BH][DK][S] bf16 (transposed)
    u16* AO = ws + 3 * NB;    // [B*S][D]   bf16 attention output

    dim3 blk(128, 1, 1);
    proj_qkv_wmma<<<dim3(64, 16, 3), blk, 0, stream>>>(q, k, v, Wq, ws);
    flash_attn_wmma<<<dim3(32, 32, 1), blk, 0, stream>>>(Qb, Kb, Vt, msk, AO);
    out_proj_wmma<<<dim3(64, 16, 1), blk, 0, stream>>>(AO, Wo, out);
}
